// Block_45578192945936
// MI455X (gfx1250) — matrix operand order repaired and checked by reading, not compiled, not run on hardware
//
#include <hip/hip_runtime.h>
#include <hip/hip_bf16.h>
#include <stdint.h>

typedef unsigned short u16;
typedef __attribute__((ext_vector_type(16))) __bf16 v16bf;
typedef __attribute__((ext_vector_type(8)))  float  v8f;
typedef __attribute__((ext_vector_type(4)))  int    i32x4;

union FragAB { v16bf v; uint4 q[2]; };

#define C_LEN 2048
#define E_DIM 1024
#define NHEAD 16
#define HDIM  64

// ---- CDNA5 async global->LDS path (probe-guarded) --------------------------
#if defined(__has_builtin)
# if __has_builtin(__builtin_amdgcn_global_load_async_to_lds_b128)
#  define HAVE_ASYNC_LDS 1
# endif
#endif
#ifndef HAVE_ASYNC_LDS
# define HAVE_ASYNC_LDS 0
#endif

__device__ __forceinline__ void async_cp16(const u16* g, u16* l) {
#if HAVE_ASYNC_LDS
    __builtin_amdgcn_global_load_async_to_lds_b128(
        (__attribute__((address_space(1))) i32x4*)g,
        (__attribute__((address_space(3))) i32x4*)l, 0, 0);
#else
    *(uint4*)l = *(const uint4*)g;
#endif
}

__device__ __forceinline__ void wait_async() {
#if HAVE_ASYNC_LDS
# if __has_builtin(__builtin_amdgcn_s_wait_asynccnt)
    __builtin_amdgcn_s_wait_asynccnt(0);
# else
    asm volatile("s_wait_asynccnt 0" ::: "memory");
# endif
#endif
}

__device__ __forceinline__ u16 f2bf(float f) {
    union { float f; unsigned u; } c; c.f = f;
    unsigned u = c.u;
    unsigned r = u + 0x7FFFu + ((u >> 16) & 1u);   // round-to-nearest-even
    return (u16)(r >> 16);
}

__device__ __forceinline__ v8f vzero8() {
    v8f z = {0.f,0.f,0.f,0.f,0.f,0.f,0.f,0.f};
    return z;
}

__device__ __forceinline__ v8f wmma_bf16(v16bf a, v16bf b, v8f c) {
    return __builtin_amdgcn_wmma_f32_16x16x32_bf16(false, a, false, b, (short)0, c, false, false);
}

// ---------------------------------------------------------------------------
// Transpose + f32->bf16 cast:  W[K,N] f32  ->  Wt[N,K] bf16
// ---------------------------------------------------------------------------
__global__ __launch_bounds__(256) void transpose_bf(const float* __restrict__ W,
                                                    u16* __restrict__ Wt,
                                                    int K, int N) {
    __shared__ float t[32][33];
    int tx = threadIdx.x, ty = threadIdx.y;
    int bx = blockIdx.x, by = blockIdx.y;      // bx tiles N, by tiles K
#pragma unroll
    for (int i = 0; i < 4; i++)
        t[ty + i*8][tx] = W[(size_t)(by*32 + ty + i*8) * N + bx*32 + tx];
    __syncthreads();
#pragma unroll
    for (int i = 0; i < 4; i++)
        Wt[(size_t)(bx*32 + ty + i*8) * K + by*32 + tx] = f2bf(t[tx][ty + i*8]);
}

// ---------------------------------------------------------------------------
// LayerNorm over E=1024, f32 in -> bf16 out. One row per 256-thread block.
// ---------------------------------------------------------------------------
__global__ __launch_bounds__(256) void ln_kernel(const float* __restrict__ x,
                                                 const float* __restrict__ g,
                                                 const float* __restrict__ be,
                                                 u16* __restrict__ y, int E) {
    int row = blockIdx.x;
    int tid = threadIdx.x;
    const float* xr = x + (size_t)row * E;
    float4 v = *(const float4*)&xr[tid*4];
    float s  = v.x + v.y + v.z + v.w;
    float sq = v.x*v.x + v.y*v.y + v.z*v.z + v.w*v.w;
#pragma unroll
    for (int m = 16; m >= 1; m >>= 1) {
        s  += __shfl_xor(s,  m);
        sq += __shfl_xor(sq, m);
    }
    __shared__ float as_[8], aq_[8];
    int wave = tid >> 5, lane = tid & 31;
    if (lane == 0) { as_[wave] = s; aq_[wave] = sq; }
    __syncthreads();
    float ts = 0.f, tq = 0.f;
#pragma unroll
    for (int i = 0; i < 8; i++) { ts += as_[i]; tq += aq_[i]; }
    float inv = 1.0f / (float)E;
    float mu  = ts * inv;
    float var = tq * inv - mu * mu;
    float rs  = rsqrtf(var + 1e-5f);
    float4 gv = *(const float4*)&g[tid*4];
    float4 bv = *(const float4*)&be[tid*4];
    ushort4 o;
    o.x = f2bf((v.x - mu) * rs * gv.x + bv.x);
    o.y = f2bf((v.y - mu) * rs * gv.y + bv.y);
    o.z = f2bf((v.z - mu) * rs * gv.z + bv.z);
    o.w = f2bf((v.w - mu) * rs * gv.w + bv.w);
    *(ushort4*)&y[(size_t)row * E + tid*4] = o;
}

// ---------------------------------------------------------------------------
// WMMA GEMM: out[M,N] = A[M,K](bf16) * Wt[N,K]^T(bf16) + bias
// Block tile 128x64, 8 waves, each wave 32x32 (2x2 WMMA tiles), K-step 32.
// Double-buffered LDS with async global->LDS prefetch of the next K tile.
// MODE 0: f32 out[M,N]
// MODE 1: bf16 out laid out [B,H,C,D]      (Q, K)
// MODE 2: bf16 out laid out [B,H,D,C]      (V, transposed)
// MODE 3: bf16 out[M,N] with ReLU          (MLP hidden)
// MODE 4: f32 out[M,N] + res[M,N]          (final)
// ---------------------------------------------------------------------------
template<int MODE>
__global__ __launch_bounds__(256) void gemm_kernel(const u16* __restrict__ A,
                                                   const u16* __restrict__ Bw,
                                                   const float* __restrict__ bias,
                                                   const float* __restrict__ res,
                                                   float* __restrict__ outF,
                                                   u16* __restrict__ outB,
                                                   int M, int N, int K) {
    __shared__ __align__(16) u16 As[2][128][40];
    __shared__ __align__(16) u16 Bs[2][64][40];

    int tid  = threadIdx.x;
    int wave = tid >> 5, lane = tid & 31;
    int half = lane >> 4, lr = lane & 15;
    int waveM = wave >> 1, waveN = wave & 1;
    int bm0 = blockIdx.y * 128, bn0 = blockIdx.x * 64;
    int m0 = waveM * 32, n0 = waveN * 32;

    v8f c[2][2];
    c[0][0] = vzero8(); c[0][1] = vzero8(); c[1][0] = vzero8(); c[1][1] = vzero8();

    int ar = tid >> 2, ac = (tid & 3) * 8;

    auto stageAB = [&](int buf, int k0) {
        async_cp16(&A [(size_t)(bm0 + ar)      * K + k0 + ac], &As[buf][ar][ac]);
        async_cp16(&A [(size_t)(bm0 + ar + 64) * K + k0 + ac], &As[buf][ar + 64][ac]);
        async_cp16(&Bw[(size_t)(bn0 + ar)      * K + k0 + ac], &Bs[buf][ar][ac]);
    };

    int nk = K >> 5;
    stageAB(0, 0);
    wait_async();
    __syncthreads();

    for (int ki = 0; ki < nk; ki++) {
        int cur = ki & 1;
        if (ki + 1 < nk) stageAB(cur ^ 1, (ki + 1) * 32);

        FragAB a[2], b[2];
#pragma unroll
        for (int t = 0; t < 2; t++) {
            const u16* ap = &As[cur][m0 + t*16 + lr][half*16];
            a[t].q[0] = *(const uint4*)ap;
            a[t].q[1] = *(const uint4*)(ap + 8);
            const u16* bp = &Bs[cur][n0 + t*16 + lr][half*16];
            b[t].q[0] = *(const uint4*)bp;
            b[t].q[1] = *(const uint4*)(bp + 8);
        }
#pragma unroll
        for (int ti = 0; ti < 2; ti++)
#pragma unroll
            for (int tj = 0; tj < 2; tj++)
                c[ti][tj] = wmma_bf16(a[ti].v, b[tj].v, c[ti][tj]);

        wait_async();
        __syncthreads();
    }

#pragma unroll
    for (int ti = 0; ti < 2; ti++)
#pragma unroll
        for (int tj = 0; tj < 2; tj++) {
            int nc = bn0 + n0 + tj*16 + lr;
            float bv = bias[nc];
#pragma unroll
            for (int r = 0; r < 8; r++) {
                int mm = bm0 + m0 + ti*16 + r + half*8;
                float val = c[ti][tj][r] + bv;
                if (MODE == 0) {
                    outF[(size_t)mm * N + nc] = val;
                } else if (MODE == 4) {
                    outF[(size_t)mm * N + nc] = val + res[(size_t)mm * N + nc];
                } else if (MODE == 3) {
                    outB[(size_t)mm * N + nc] = f2bf(fmaxf(val, 0.0f));
                } else {
                    int hh = nc >> 6, dd = nc & 63;
                    int bb = mm >> 11, cc = mm & 2047;
                    size_t base = (size_t)(bb * NHEAD + hh);
                    if (MODE == 1) outB[(base * C_LEN + cc) * HDIM + dd] = f2bf(val);
                    else           outB[(base * HDIM + dd) * C_LEN + cc] = f2bf(val);
                }
            }
        }
}

// ---------------------------------------------------------------------------
// Flash attention, causal. One block = (b,h) x 128 q-rows; 8 waves x 16 rows.
// Qb,Kb: bf16 [B,H,C,D]; Vb: bf16 [B,H,D,C]; out Y = inputs + attn, f32 [B,C,E].
// K/V tiles double-buffered in LDS with async prefetch.
// ---------------------------------------------------------------------------
__global__ __launch_bounds__(256) void attn_kernel(const u16* __restrict__ Qb,
                                                   const u16* __restrict__ Kb,
                                                   const u16* __restrict__ Vb,
                                                   const float* __restrict__ inp,
                                                   float* __restrict__ Y) {
    __shared__ __align__(16) u16 Ks [2][32][72];
    __shared__ __align__(16) u16 VsT[2][64][40];
    __shared__ __align__(16) u16 Ps [8][16][40];

    int tid  = threadIdx.x;
    int wave = tid >> 5, lane = tid & 31;
    int half = lane >> 4, lr = lane & 15;
    int bh = blockIdx.y;
    int b = bh >> 4, h = bh & 15;
    int qblock = blockIdx.x * 128;

    // Q fragments for this wave's 16 rows (A-matrix layout, D split into two K=32 chunks)
    const u16* qp = Qb + ((size_t)bh * C_LEN + (qblock + wave*16 + lr)) * HDIM;
    FragAB qa0, qa1;
    qa0.q[0] = *(const uint4*)(qp + half*16);
    qa0.q[1] = *(const uint4*)(qp + 8 + half*16);
    qa1.q[0] = *(const uint4*)(qp + 32 + half*16);
    qa1.q[1] = *(const uint4*)(qp + 40 + half*16);

    v8f o[4];
    o[0] = vzero8(); o[1] = vzero8(); o[2] = vzero8(); o[3] = vzero8();
    float mrow[8], lrow[8];
#pragma unroll
    for (int r = 0; r < 8; r++) { mrow[r] = -1e30f; lrow[r] = 0.f; }

    auto stageKV = [&](int buf, int kb) {
        int r  = tid >> 3, c8 = (tid & 7) * 8;
        async_cp16(&Kb[((size_t)bh * C_LEN + kb + r) * HDIM + c8], &Ks[buf][r][c8]);
        int vr = tid >> 2, vc8 = (tid & 3) * 8;
        async_cp16(&Vb[((size_t)bh * HDIM + vr) * C_LEN + kb + vc8], &VsT[buf][vr][vc8]);
    };

    int nkb = (qblock >> 5) + 4;   // key blocks of 32, covering keys < qblock+128
    stageKV(0, 0);
    wait_async();
    __syncthreads();

    for (int kbi = 0; kbi < nkb; kbi++) {
        int kb = kbi * 32;
        int cur = kbi & 1;
        if (kbi + 1 < nkb) stageKV(cur ^ 1, (kbi + 1) * 32);

        // ---- S = Q K^T (two 16x16 n-tiles) ----
        float s0[8], s1[8];
        {
            v8f acc0 = vzero8(), acc1 = vzero8();
            FragAB kf;
            const u16* kp;
            kp = &Ks[cur][lr][half*16];
            kf.q[0] = *(const uint4*)kp; kf.q[1] = *(const uint4*)(kp + 8);
            acc0 = wmma_bf16(qa0.v, kf.v, acc0);
            kp = &Ks[cur][lr][32 + half*16];
            kf.q[0] = *(const uint4*)kp; kf.q[1] = *(const uint4*)(kp + 8);
            acc0 = wmma_bf16(qa1.v, kf.v, acc0);
            kp = &Ks[cur][16 + lr][half*16];
            kf.q[0] = *(const uint4*)kp; kf.q[1] = *(const uint4*)(kp + 8);
            acc1 = wmma_bf16(qa0.v, kf.v, acc1);
            kp = &Ks[cur][16 + lr][32 + half*16];
            kf.q[0] = *(const uint4*)kp; kf.q[1] = *(const uint4*)(kp + 8);
            acc1 = wmma_bf16(qa1.v, kf.v, acc1);

            int qbase = qblock + wave*16 + half*8;
#pragma unroll
            for (int r = 0; r < 8; r++) {
                int q = qbase + r;
                s0[r] = (kb + lr      <= q) ? acc0[r] * 0.125f : -1e30f;
                s1[r] = (kb + 16 + lr <= q) ? acc1[r] * 0.125f : -1e30f;
            }
        }

        // ---- online softmax (row = element r within a lane half) ----
        float corr[8];
#pragma unroll
        for (int r = 0; r < 8; r++) {
            float mv = fmaxf(s0[r], s1[r]);
            mv = fmaxf(mv, __shfl_xor(mv, 8));
            mv = fmaxf(mv, __shfl_xor(mv, 4));
            mv = fmaxf(mv, __shfl_xor(mv, 2));
            mv = fmaxf(mv, __shfl_xor(mv, 1));
            float nm = fmaxf(mrow[r], mv);
            corr[r]  = __expf(mrow[r] - nm);
            mrow[r]  = nm;
            s0[r] = __expf(s0[r] - nm);
            s1[r] = __expf(s1[r] - nm);
            float rs = s0[r] + s1[r];
            rs += __shfl_xor(rs, 8);
            rs += __shfl_xor(rs, 4);
            rs += __shfl_xor(rs, 2);
            rs += __shfl_xor(rs, 1);
            lrow[r] = lrow[r] * corr[r] + rs;
        }
#pragma unroll
        for (int dt = 0; dt < 4; dt++)
#pragma unroll
            for (int r = 0; r < 8; r++) o[dt][r] *= corr[r];

        // ---- re-layout P (C-frag -> A-frag) through LDS ----
#pragma unroll
        for (int r = 0; r < 8; r++) {
            Ps[wave][r + half*8][lr]      = f2bf(s0[r]);
            Ps[wave][r + half*8][16 + lr] = f2bf(s1[r]);
        }
        asm volatile("s_wait_dscnt 0" ::: "memory");
        FragAB pf;
        {
            const u16* pp = &Ps[wave][lr][half*16];
            pf.q[0] = *(const uint4*)pp;
            pf.q[1] = *(const uint4*)(pp + 8);
        }

        // ---- O += P V ----
#pragma unroll
        for (int dt = 0; dt < 4; dt++) {
            FragAB vf;
            const u16* vp = &VsT[cur][dt*16 + lr][half*16];
            vf.q[0] = *(const uint4*)vp;
            vf.q[1] = *(const uint4*)(vp + 8);
            o[dt] = wmma_bf16(pf.v, vf.v, o[dt]);
        }

        wait_async();
        __syncthreads();
    }

    // ---- epilogue: Y = inputs + O / l ----
#pragma unroll
    for (int dt = 0; dt < 4; dt++)
#pragma unroll
        for (int r = 0; r < 8; r++) {
            int q = qblock + wave*16 + r + half*8;
            int e = h * HDIM + dt*16 + lr;
            size_t idx = ((size_t)(b * C_LEN + q)) * E_DIM + e;
            Y[idx] = inp[idx] + o[dt][r] / lrow[r];
        }
}

// ---------------------------------------------------------------------------
extern "C" void kernel_launch(void* const* d_in, const int* in_sizes, int n_in,
                              void* d_out, int out_size, void* d_ws, size_t ws_size,
                              hipStream_t stream) {
    (void)in_sizes; (void)n_in; (void)out_size; (void)ws_size;
    const float* inputs = (const float*)d_in[0];
    const float* Wq  = (const float*)d_in[1];
    const float* bq  = (const float*)d_in[2];
    const float* Wk  = (const float*)d_in[3];
    const float* bk  = (const float*)d_in[4];
    const float* Wv  = (const float*)d_in[5];
    const float* bv  = (const float*)d_in[6];
    const float* g1  = (const float*)d_in[7];
    const float* be1 = (const float*)d_in[8];
    const float* g2  = (const float*)d_in[9];
    const float* be2 = (const float*)d_in[10];
    const float* W1  = (const float*)d_in[11];
    const float* bm1 = (const float*)d_in[12];
    const float* W2  = (const float*)d_in[13];
    const float* bm2 = (const float*)d_in[14];
    float* out = (float*)d_out;

    const int M = 8192, E = 1024, F = 4096;

    char* w = (char*)d_ws;
    size_t off = 0;
    auto carve = [&](size_t bytes) -> char* {
        char* p = w + off;
        off += (bytes + 255) & ~(size_t)255;
        return p;
    };
    u16*   WqT = (u16*)  carve((size_t)E * E * 2);
    u16*   WkT = (u16*)  carve((size_t)E * E * 2);
    u16*   WvT = (u16*)  carve((size_t)E * E * 2);
    u16*   W1T = (u16*)  carve((size_t)E * F * 2);
    u16*   W2T = (u16*)  carve((size_t)F * E * 2);
    u16*   X1  = (u16*)  carve((size_t)M * E * 2);
    u16*   Qb  = (u16*)  carve((size_t)M * E * 2);
    u16*   Kb  = (u16*)  carve((size_t)M * E * 2);
    u16*   Vb  = (u16*)  carve((size_t)M * E * 2);
    float* Y   = (float*)carve((size_t)M * E * 4);
    u16*   X2  = (u16*)  carve((size_t)M * E * 2);
    u16*   Hb  = (u16*)  carve((size_t)M * F * 2);

    dim3 tb(32, 8);
    transpose_bf<<<dim3(E/32, E/32), tb, 0, stream>>>(Wq, WqT, E, E);
    transpose_bf<<<dim3(E/32, E/32), tb, 0, stream>>>(Wk, WkT, E, E);
    transpose_bf<<<dim3(E/32, E/32), tb, 0, stream>>>(Wv, WvT, E, E);
    transpose_bf<<<dim3(F/32, E/32), tb, 0, stream>>>(W1, W1T, E, F);
    transpose_bf<<<dim3(E/32, F/32), tb, 0, stream>>>(W2, W2T, F, E);

    ln_kernel<<<M, 256, 0, stream>>>(inputs, g1, be1, X1, E);

    gemm_kernel<1><<<dim3(E/64, M/128), 256, 0, stream>>>(X1, WqT, bq, nullptr, nullptr, Qb, M, E, E);
    gemm_kernel<1><<<dim3(E/64, M/128), 256, 0, stream>>>(X1, WkT, bk, nullptr, nullptr, Kb, M, E, E);
    gemm_kernel<2><<<dim3(E/64, M/128), 256, 0, stream>>>(X1, WvT, bv, nullptr, nullptr, Vb, M, E, E);

    attn_kernel<<<dim3(C_LEN/128, 4*NHEAD), 256, 0, stream>>>(Qb, Kb, Vb, inputs, Y);

    ln_kernel<<<M, 256, 0, stream>>>(Y, g2, be2, X2, E);

    gemm_kernel<3><<<dim3(F/64, M/128), 256, 0, stream>>>(X2, W1T, bm1, nullptr, nullptr, Hb, M, F, E);
    gemm_kernel<4><<<dim3(E/64, M/128), 256, 0, stream>>>(Hb, W2T, bm2, Y, out, nullptr, M, E, F);
}
